// PointNetSaModule_75196287418834
// MI455X (gfx1250) — matrix operand order repaired and checked by reading, not compiled, not run on hardware
//
#include <hip/hip_runtime.h>
#include <hip/hip_bf16.h>

// ---------------------------------------------------------------------------
// PointNet++ Set Abstraction module for MI455X (gfx1250, wave32, WMMA).
// Problem sizes fixed by the reference's setup_inputs():
//   B=8, N=8192, C=64, NPOINT(S)=2048, K=16, MLP dims 67->64->64->128.
// MLP GEMMs are HBM-bound => f16 activations + v_wmma_f32_16x16x32_f16
// (f32 accumulate), single-pass fused BN/ReLU stages, deterministic
// fixed-order BN reductions (no float atomics) for graph-replay stability.
// Workspace use: ~178 MB of d_ws.
// ---------------------------------------------------------------------------

#define B_ 8
#define N_ 8192
#define C_ 64
#define S_ 2048           // NPOINT
#define K_ 16
#define MTOT (B_ * S_ * K_)   // 262144 rows for the MLP GEMMs
#define STATS_BLOCKS 1024

typedef __attribute__((ext_vector_type(16))) _Float16 v16h;
typedef __attribute__((ext_vector_type(8)))  _Float16 v8h;
typedef __attribute__((ext_vector_type(8)))  float    v8f;

// ---------------------------------------------------------------------------
// 1) Furthest point sampling: one block per batch, 1024 threads, 8 pts/thread.
//    Points + running min-dist live in registers; block argmax via wave
//    shuffles + LDS. Tie-break = lowest index (matches jnp.argmax).
// ---------------------------------------------------------------------------
__global__ void fps_kernel(const float* __restrict__ xyz, int* __restrict__ idx_out) {
  const int b = blockIdx.x;
  const int tid = threadIdx.x;              // 1024 threads
  const float* p = xyz + (size_t)b * N_ * 3;

  __shared__ float s_red[32];
  __shared__ int   s_redi[32];

  float px[8], py[8], pz[8], mind[8];
#pragma unroll
  for (int j = 0; j < 8; ++j) {
    int n = tid + j * 1024;
    px[j] = p[n * 3 + 0];
    py[j] = p[n * 3 + 1];
    pz[j] = p[n * 3 + 2];
    mind[j] = 1e10f;
  }

  int farthest = 0;
  for (int i = 0; i < S_; ++i) {
    if (tid == 0) idx_out[(size_t)b * S_ + i] = farthest;
    // broadcast centroid read (same address for all lanes -> cached)
    float cx = p[farthest * 3 + 0];
    float cy = p[farthest * 3 + 1];
    float cz = p[farthest * 3 + 2];

    float best = -1.0f;
    int bestn = 0x7fffffff;
#pragma unroll
    for (int j = 0; j < 8; ++j) {
      float dx = px[j] - cx, dy = py[j] - cy, dz = pz[j] - cz;
      float d = dx * dx + dy * dy + dz * dz;
      mind[j] = fminf(mind[j], d);
      if (mind[j] > best) { best = mind[j]; bestn = tid + j * 1024; }
    }
    // wave32 argmax (ties -> lowest index)
    for (int off = 16; off; off >>= 1) {
      float od = __shfl_xor(best, off);
      int   on = __shfl_xor(bestn, off);
      if (od > best || (od == best && on < bestn)) { best = od; bestn = on; }
    }
    if ((tid & 31) == 0) { s_red[tid >> 5] = best; s_redi[tid >> 5] = bestn; }
    __syncthreads();
    if (tid < 32) {
      best = s_red[tid]; bestn = s_redi[tid];
      for (int off = 16; off; off >>= 1) {
        float od = __shfl_xor(best, off);
        int   on = __shfl_xor(bestn, off);
        if (od > best || (od == best && on < bestn)) { best = od; bestn = on; }
      }
      if (tid == 0) s_redi[0] = bestn;
    }
    __syncthreads();
    farthest = s_redi[0];
    __syncthreads();   // keep s_redi[0] stable until everyone has read it
  }
}

// ---------------------------------------------------------------------------
// 2) Gather sampled xyz/label + emit idx (outputs 1, 2, 4 of the tuple).
// ---------------------------------------------------------------------------
__global__ void gather_sampled(const float* __restrict__ xyz, const float* __restrict__ label,
                               const int* __restrict__ idx,
                               float* __restrict__ oxyz, float* __restrict__ olabel,
                               int* __restrict__ oidx) {
  int t = blockIdx.x * blockDim.x + threadIdx.x;
  if (t >= B_ * S_) return;
  int b = t / S_;
  int n = idx[t];
#pragma unroll
  for (int c = 0; c < 3; ++c) {
    oxyz[(size_t)t * 3 + c]   = xyz[((size_t)b * N_ + n) * 3 + c];
    olabel[(size_t)t * 3 + c] = label[((size_t)b * N_ + n) * 3 + c];
  }
  oidx[t] = n;
}

// ---------------------------------------------------------------------------
// 3) Brute-force kNN: one wave32 per query. Per-lane register insertion-sort
//    top-16 over a strided slice of the 8192 points, then 16 rounds of
//    wave-wide lexicographic argmin to merge. Distances use the reference's
//    expanded |q|^2+|p|^2-2q.p formula; stable ordering on ties.
// ---------------------------------------------------------------------------
__global__ void knn_kernel(const float* __restrict__ xyz, const float* __restrict__ new_xyz,
                           int* __restrict__ knn) {
  int wave = blockIdx.x * (blockDim.x >> 5) + (threadIdx.x >> 5);
  int lane = threadIdx.x & 31;
  if (wave >= B_ * S_) return;
  int b = wave / S_;
  const float* p = xyz + (size_t)b * N_ * 3;
  float qx = new_xyz[(size_t)wave * 3 + 0];
  float qy = new_xyz[(size_t)wave * 3 + 1];
  float qz = new_xyz[(size_t)wave * 3 + 2];
  float qq = qx * qx + qy * qy + qz * qz;

  float dk[K_]; int ik[K_];
#pragma unroll
  for (int j = 0; j < K_; ++j) { dk[j] = 3.4e38f; ik[j] = 0x7fffffff; }

  for (int n = lane; n < N_; n += 32) {
    float x = p[n * 3 + 0], y = p[n * 3 + 1], z = p[n * 3 + 2];
    float d = qq + (x * x + y * y + z * z) - 2.0f * (qx * x + qy * y + qz * z);
    float nd = d; int ni = n;
#pragma unroll
    for (int j = 0; j < K_; ++j) {            // sorted insertion, static indices
      bool sw = (nd < dk[j]) || (nd == dk[j] && ni < ik[j]);
      float td = sw ? dk[j] : nd;  int ti = sw ? ik[j] : ni;
      dk[j] = sw ? nd : dk[j];     ik[j] = sw ? ni : ik[j];
      nd = td; ni = ti;
    }
  }
#pragma unroll
  for (int r = 0; r < K_; ++r) {
    float best = dk[0]; int bi = ik[0]; int bl = lane;
    for (int off = 16; off; off >>= 1) {
      float od = __shfl_xor(best, off);
      int   oi = __shfl_xor(bi, off);
      int   ol = __shfl_xor(bl, off);
      if (od < best || (od == best && oi < bi)) { best = od; bi = oi; bl = ol; }
    }
    if (lane == 0) knn[(size_t)wave * K_ + r] = bi;
    if (lane == bl) {                          // winner pops its head
#pragma unroll
      for (int j = 0; j < K_ - 1; ++j) { dk[j] = dk[j + 1]; ik[j] = ik[j + 1]; }
      dk[K_ - 1] = 3.4e38f; ik[K_ - 1] = 0x7fffffff;
    }
  }
}

// ---------------------------------------------------------------------------
// 4) Build layer-0 features as f16, K-padded 67 -> 96 (3 x 32 WMMA K-steps).
// ---------------------------------------------------------------------------
__global__ void build_features(const float* __restrict__ xyz, const float* __restrict__ points,
                               const float* __restrict__ new_xyz, const int* __restrict__ knn,
                               _Float16* __restrict__ X0) {
  size_t t = (size_t)blockIdx.x * blockDim.x + threadIdx.x;
  if (t >= (size_t)MTOT * 96) return;
  int col = (int)(t % 96);
  size_t row = t / 96;
  size_t bs = row / K_;            // b*S + s
  int b = (int)(bs / S_);
  int n = knn[row];
  float v = 0.0f;
  if (col < 3)            v = xyz[((size_t)b * N_ + n) * 3 + col] - new_xyz[bs * 3 + col];
  else if (col < 3 + C_)  v = points[((size_t)b * N_ + n) * C_ + (col - 3)];
  X0[t] = (_Float16)v;
}

// ---------------------------------------------------------------------------
// 5) Pack W (f32 [Cin,Cout]) into the CDNA5 B-fragment layout (f16), K padded.
//    Per ISA 7.12.2, 16-bit B 32x16: lane = (half<<4)|col, element e -> K =
//    half*8 + (e&7) + (e>>3)*16. Fragment for (kt,nt) stored contiguously: 32 lanes x 16 f16.
// ---------------------------------------------------------------------------
__global__ void pack_w(const float* __restrict__ W, _Float16* __restrict__ Wp,
                       int Cin, int Cout, int KT) {
  int t = blockIdx.x * blockDim.x + threadIdx.x;
  int NT = Cout >> 4;
  int total = KT * NT * 512;
  if (t >= total) return;
  int e    = t & 15;
  int lane = (t >> 4) & 31;
  int rest = t >> 9;
  int nt = rest % NT;
  int kt = rest / NT;
  int Kg = kt * 32 + (lane >> 4) * 8 + (e & 7) + (e >> 3) * 16;
  int c  = nt * 16 + (lane & 15);
  _Float16 v = (_Float16)0.0f;
  if (Kg < Cin) v = (_Float16)W[(size_t)Kg * Cout + c];
  Wp[t] = v;
}

// ---------------------------------------------------------------------------
// 6) WMMA GEMM: Y[M,Cout] = X[M,Kpad] * W (f16 in, f32 accumulate).
//    One wave computes a 16 x (NT*16) strip; A fragment reused across NT.
//    A layout per ISA 7.12.2 (16-bit A 16x32): lane half/row; elements 0..7 =
//    K kbase+half*8.., elements 8..15 = K kbase+16+half*8.. -> two b128 loads.
// ---------------------------------------------------------------------------
template <int NT, int KT>
__global__ void wmma_gemm(const _Float16* __restrict__ X, const _Float16* __restrict__ Wp,
                          float* __restrict__ Y, int tilesM) {
  constexpr int Kpad = KT * 32;
  constexpr int Cout = NT * 16;
  int mt = blockIdx.x * (blockDim.x >> 5) + (threadIdx.x >> 5);
  if (mt >= tilesM) return;
  int lane = threadIdx.x & 31;
  int half = lane >> 4;
  int r    = lane & 15;

  const _Float16* xrow = X + (size_t)(mt * 16 + r) * Kpad;

  v8f zero = {};
  v8f acc[NT];
#pragma unroll
  for (int i = 0; i < NT; ++i) acc[i] = zero;

#pragma unroll
  for (int kt = 0; kt < KT; ++kt) {
    union { v16h v; v8h h[2]; } a;
    a.h[0] = *(const v8h*)(xrow + kt * 32 + half * 8);        // K = kbase+half*8 .. +7
    a.h[1] = *(const v8h*)(xrow + kt * 32 + 16 + half * 8);   // K = kbase+16+half*8 ..
#pragma unroll
    for (int nt = 0; nt < NT; ++nt) {
      v16h bfrag = *(const v16h*)(Wp + ((size_t)(kt * NT + nt) * 32 + lane) * 16);
      acc[nt] = __builtin_amdgcn_wmma_f32_16x16x32_f16(
          /*neg_a=*/false, a.v, /*neg_b=*/false, bfrag,
          /*c_mod=*/(short)0, acc[nt], /*reuse_a=*/false, /*reuse_b=*/false);
    }
  }

  // C/D layout: VGPR v -> row v + 8*half; col = nt*16 + (lane&15)
  float* ybase = Y + (size_t)(mt * 16) * Cout + r;
#pragma unroll
  for (int nt = 0; nt < NT; ++nt) {
#pragma unroll
    for (int v = 0; v < 8; ++v)
      ybase[(size_t)(v + 8 * half) * Cout + nt * 16] = acc[nt][v];
  }
}

// ---------------------------------------------------------------------------
// 7) BN batch statistics: per-block partial sums (fully coalesced: thread tid
//    owns channel tid%Cout), then a fixed-order reduce+finalize kernel.
//    No float atomics -> bitwise deterministic across graph replays.
// ---------------------------------------------------------------------------
__global__ void stats_partial(const float* __restrict__ Y, float* __restrict__ part,
                              int Cout, int Mrows) {
  __shared__ float ls[256], ls2[256];
  int tid = threadIdx.x;
  int c = tid & (Cout - 1);
  int r0 = tid / Cout;
  int rstep = 256 / Cout;
  int rowsPerBlock = (Mrows + gridDim.x - 1) / gridDim.x;
  int rbeg = blockIdx.x * rowsPerBlock;
  int rend = rbeg + rowsPerBlock; if (rend > Mrows) rend = Mrows;
  float s = 0.0f, s2 = 0.0f;
  for (int r = rbeg + r0; r < rend; r += rstep) {
    float v = Y[(size_t)r * Cout + c];
    s += v; s2 += v * v;
  }
  ls[tid] = s; ls2[tid] = s2;
  __syncthreads();
  for (int off = 128; off >= Cout; off >>= 1) {
    if (tid < off) { ls[tid] += ls[tid + off]; ls2[tid] += ls2[tid + off]; }
    __syncthreads();
  }
  if (tid < Cout) {
    part[(size_t)blockIdx.x * 2 * Cout + tid]        = ls[tid];
    part[(size_t)blockIdx.x * 2 * Cout + Cout + tid] = ls2[tid];
  }
}

__global__ void reduce_finalize(const float* __restrict__ part, int nparts, int Cout,
                                float invM, float* __restrict__ mu, float* __restrict__ rstd) {
  __shared__ float tot[256];
  int tid = threadIdx.x;
  if (tid < 2 * Cout) {
    float s = 0.0f;
    for (int i = 0; i < nparts; ++i) s += part[(size_t)i * 2 * Cout + tid];
    tot[tid] = s;
  }
  __syncthreads();
  if (tid < Cout) {
    float m = tot[tid] * invM;
    float v = tot[Cout + tid] * invM - m * m;
    mu[tid] = m;
    rstd[tid] = rsqrtf(v + 1e-5f);
  }
}

// ---------------------------------------------------------------------------
// 8) Fused BN + ReLU -> f16 (feeds the next GEMM).  Conv bias cancels exactly
//    through batch-stat BN, so it is never added.
// ---------------------------------------------------------------------------
__global__ void bn_relu_f16(const float* __restrict__ Y, const float* __restrict__ mu,
                            const float* __restrict__ rstd, const float* __restrict__ g,
                            const float* __restrict__ bt, _Float16* __restrict__ Xo,
                            int Cout, size_t total) {
  size_t t = (size_t)blockIdx.x * blockDim.x + threadIdx.x;
  if (t >= total) return;
  int c = (int)(t & (size_t)(Cout - 1));
  float v = (Y[t] - mu[c]) * rstd[c] * g[c] + bt[c];
  Xo[t] = (_Float16)fmaxf(v, 0.0f);
}

// ---------------------------------------------------------------------------
// 9) Final fused BN + ReLU + max over K neighbors -> new_points [B,S,128].
// ---------------------------------------------------------------------------
__global__ void bn_relu_maxpool(const float* __restrict__ Y, const float* __restrict__ mu,
                                const float* __restrict__ rstd, const float* __restrict__ g,
                                const float* __restrict__ bt, float* __restrict__ out) {
  size_t t = (size_t)blockIdx.x * blockDim.x + threadIdx.x;
  if (t >= (size_t)B_ * S_ * 128) return;
  int c = (int)(t % 128);
  size_t bs = t / 128;
  float m = mu[c], rs = rstd[c], gg = g[c], bb = bt[c];
  float best = 0.0f;                         // ReLU outputs are >= 0
#pragma unroll
  for (int k = 0; k < K_; ++k) {
    float v = (Y[(bs * K_ + k) * (size_t)128 + c] - m) * rs * gg + bb;
    best = fmaxf(best, fmaxf(v, 0.0f));
  }
  out[t] = best;
}

// ---------------------------------------------------------------------------
// Host-side orchestration
// ---------------------------------------------------------------------------
extern "C" void kernel_launch(void* const* d_in, const int* in_sizes, int n_in,
                              void* d_out, int out_size, void* d_ws, size_t ws_size,
                              hipStream_t stream) {
  (void)in_sizes; (void)n_in; (void)out_size; (void)ws_size;

  const float* xyz    = (const float*)d_in[0];
  const float* label  = (const float*)d_in[1];
  const float* points = (const float*)d_in[2];
  const float* w0  = (const float*)d_in[3];
  // b0/b1/b2 (d_in[4],[8],[12]) provably cancel through batch-stat BN.
  const float* g0  = (const float*)d_in[5];
  const float* bt0 = (const float*)d_in[6];
  const float* w1  = (const float*)d_in[7];
  const float* g1  = (const float*)d_in[9];
  const float* bt1 = (const float*)d_in[10];
  const float* w2  = (const float*)d_in[11];
  const float* g2  = (const float*)d_in[13];
  const float* bt2 = (const float*)d_in[14];

  // Output layout (flat, return order): new_xyz | new_label | new_points | idx
  float* out       = (float*)d_out;
  float* out_xyz   = out;                          // B*S*3   = 49152
  float* out_label = out + (size_t)B_ * S_ * 3;    // B*S*3
  float* out_pts   = out + (size_t)B_ * S_ * 6;    // B*S*128 = 2097152
  int*   out_idx   = (int*)(out + (size_t)B_ * S_ * 6 + (size_t)B_ * S_ * 128);

  // Workspace carve-up (256B aligned), total ~178 MB.
  char* base = (char*)d_ws;
  size_t o = 0;
  auto alloc = [&](size_t bytes) -> void* {
    o = (o + 255) & ~(size_t)255;
    void* p = base + o;
    o += bytes;
    return p;
  };
  int*      ws_idx = (int*)alloc((size_t)B_ * S_ * sizeof(int));
  int*      ws_knn = (int*)alloc((size_t)MTOT * sizeof(int));
  _Float16* Wp0    = (_Float16*)alloc((size_t)3 * 4 * 512 * sizeof(_Float16));
  _Float16* Wp1    = (_Float16*)alloc((size_t)2 * 4 * 512 * sizeof(_Float16));
  _Float16* Wp2    = (_Float16*)alloc((size_t)2 * 8 * 512 * sizeof(_Float16));
  float*    part   = (float*)alloc((size_t)STATS_BLOCKS * 256 * sizeof(float));
  float*    mu     = (float*)alloc(128 * sizeof(float));
  float*    rstd   = (float*)alloc(128 * sizeof(float));
  _Float16* Xbuf   = (_Float16*)alloc((size_t)MTOT * 96 * sizeof(_Float16));  // 48 MB (reused [M,64])
  float*    Ybuf   = (float*)alloc((size_t)MTOT * 128 * sizeof(float));       // 128 MB

  const float invM = 1.0f / (float)MTOT;
  const int tilesM = MTOT / 16;            // 16384
  const int gemmBlocks = tilesM / 8;       // 8 waves per 256-thread block

  // --- sampling + grouping ---
  fps_kernel<<<B_, 1024, 0, stream>>>(xyz, ws_idx);
  gather_sampled<<<(B_ * S_ + 255) / 256, 256, 0, stream>>>(xyz, label, ws_idx,
                                                            out_xyz, out_label, out_idx);
  knn_kernel<<<(B_ * S_) / 8, 256, 0, stream>>>(xyz, out_xyz, ws_knn);
  build_features<<<(int)(((size_t)MTOT * 96 + 255) / 256), 256, 0, stream>>>(
      xyz, points, out_xyz, ws_knn, Xbuf);

  // --- weight packing into B-fragment layout ---
  pack_w<<<(3 * 4 * 512 + 255) / 256, 256, 0, stream>>>(w0, Wp0, 67, 64, 3);
  pack_w<<<(2 * 4 * 512 + 255) / 256, 256, 0, stream>>>(w1, Wp1, 64, 64, 2);
  pack_w<<<(2 * 8 * 512 + 255) / 256, 256, 0, stream>>>(w2, Wp2, 64, 128, 2);

  // --- layer 0: [M,96(f16)] x [96,64] ---
  wmma_gemm<4, 3><<<gemmBlocks, 256, 0, stream>>>(Xbuf, Wp0, Ybuf, tilesM);
  stats_partial<<<STATS_BLOCKS, 256, 0, stream>>>(Ybuf, part, 64, MTOT);
  reduce_finalize<<<1, 256, 0, stream>>>(part, STATS_BLOCKS, 64, invM, mu, rstd);
  bn_relu_f16<<<(int)(((size_t)MTOT * 64 + 255) / 256), 256, 0, stream>>>(
      Ybuf, mu, rstd, g0, bt0, Xbuf, 64, (size_t)MTOT * 64);

  // --- layer 1: [M,64] x [64,64] ---
  wmma_gemm<4, 2><<<gemmBlocks, 256, 0, stream>>>(Xbuf, Wp1, Ybuf, tilesM);
  stats_partial<<<STATS_BLOCKS, 256, 0, stream>>>(Ybuf, part, 64, MTOT);
  reduce_finalize<<<1, 256, 0, stream>>>(part, STATS_BLOCKS, 64, invM, mu, rstd);
  bn_relu_f16<<<(int)(((size_t)MTOT * 64 + 255) / 256), 256, 0, stream>>>(
      Ybuf, mu, rstd, g1, bt1, Xbuf, 64, (size_t)MTOT * 64);

  // --- layer 2: [M,64] x [64,128], BN+ReLU fused into max-pool ---
  wmma_gemm<8, 2><<<gemmBlocks, 256, 0, stream>>>(Xbuf, Wp2, Ybuf, tilesM);
  stats_partial<<<STATS_BLOCKS, 256, 0, stream>>>(Ybuf, part, 128, MTOT);
  reduce_finalize<<<1, 256, 0, stream>>>(part, STATS_BLOCKS, 128, invM, mu, rstd);
  bn_relu_maxpool<<<(int)(((size_t)B_ * S_ * 128 + 255) / 256), 256, 0, stream>>>(
      Ybuf, mu, rstd, g2, bt2, out_pts);
}
